// TCN_88433376625047
// MI455X (gfx1250) — matrix operand order repaired and checked by reading, not compiled, not run on hardware
//
#include <hip/hip_runtime.h>

#define T_LEN 4000
#define EPSF 1e-8f

typedef __attribute__((ext_vector_type(16))) __bf16 v16bf;
typedef __attribute__((ext_vector_type(8)))  float  v8f;

// ---------- helpers ----------

__device__ __forceinline__ unsigned short f2bf(float f) {
  union { float f; unsigned u; } v; v.f = f;
  unsigned u = v.u;
  u += 0x7FFFu + ((u >> 16) & 1u);   // round-to-nearest-even
  return (unsigned short)(u >> 16);
}

// A-matrix 16x32 bf16 swizzle (ISA 7.12.2): lane = 16*((k>>3)&1) + m,
// half-idx = (k&7) + 8*(k>>4). Fragment = 16 contiguous halfs per lane.
__device__ __forceinline__ int slotA(int kk, int m) {
  return (((((kk >> 3) & 1) << 4) + m) << 4) + ((kk & 7) | ((kk >> 4) << 3));
}
// B-matrix 32x16 bf16 swizzle (same k order as A): lane = 16*((k>>3)&1) + n,
// half-idx = (k&7) + 8*(k>>4).
__device__ __forceinline__ int slotB(int kk, int n) {
  return (((((kk >> 3) & 1) << 4) + n) << 4) + ((kk & 7) | ((kk >> 4) << 3));
}

__device__ __forceinline__ v8f wmma_bf(v16bf a, v16bf b, v8f c) {
  return __builtin_amdgcn_wmma_f32_16x16x32_bf16(false, a, false, b,
                                                 (short)0, c, false, false);
}

__device__ __forceinline__ void red2_atomic(float s, float s2, float* red,
                                            float* oA, float* oB) {
  int tid = threadIdx.x;
  red[tid] = s; red[256 + tid] = s2;
  __syncthreads();
  for (int off = 128; off > 0; off >>= 1) {
    if (tid < off) {
      red[tid] += red[tid + off];
      red[256 + tid] += red[256 + tid + off];
    }
    __syncthreads();
  }
  if (tid == 0) { atomicAdd(oA, red[0]); atomicAdd(oB, red[256]); }
}

// ---------- kernel: pre-swizzle f32 weights into WMMA-fragment bf16 layout ----------
// Matrix [M][K] row-major -> chunks of [128 rows][KCH cols], ordered
// (m-chunk major, then k-chunk), each chunk = [mtile8][kt=KCH/32][32 lanes][16].

__global__ __launch_bounds__(256) void prep_w(const float* __restrict__ W,
                                              unsigned short* __restrict__ out,
                                              int M, int K, int KCH) {
  int idx = blockIdx.x * 256 + threadIdx.x;
  if (idx >= M * K) return;
  int m = idx / K, k = idx - m * K;
  int mc = m >> 7, kc = k / KCH;
  int ml = m & 127, kl = k - kc * KCH;
  int ktiles = KCH >> 5;
  int nkc = K / KCH;
  size_t chunkhalfs = (size_t)128 * KCH;
  out[(size_t)(mc * nkc + kc) * chunkhalfs +
      ((((ml >> 4) * ktiles) + (kl >> 5)) << 9) + slotA(kl & 31, ml & 15)] =
      f2bf(W[(size_t)m * K + k]);
}

// ---------- kernel: per-sample sum / sumsq over [C*T] ----------

__global__ __launch_bounds__(256) void stats_sum(const float* __restrict__ X,
                                                 int per_b,
                                                 float* __restrict__ stats) {
  __shared__ float red[512];
  const int tid = threadIdx.x, b = blockIdx.z;
  const float* base = X + (size_t)b * per_b;
  float s = 0.f, s2 = 0.f;
  for (int i = blockIdx.x * 256 + tid; i < per_b; i += gridDim.x * 256) {
    float v = base[i]; s += v; s2 += v * v;
  }
  red2_atomic(s, s2, red, &stats[b * 2], &stats[b * 2 + 1]);
}

// ---------- kernel: Z = prelu(W[512,128] @ X[128,T] + bias), GN1 stats ----------

__global__ __launch_bounds__(256) void gemm_pw(const float* __restrict__ X,
                                               const unsigned short* __restrict__ Wsw,
                                               const float* __restrict__ bias,
                                               const float* __restrict__ p1,
                                               float* __restrict__ Z,
                                               float* __restrict__ stats) {
  __shared__ alignas(32) unsigned short lA[128 * 128];  // [mtile8][kt4][512]
  __shared__ alignas(32) unsigned short lB[128 * 32];   // [kt4][ntile2][512]
  __shared__ float red[512];

  const int tid = threadIdx.x;
  const int b = blockIdx.z;
  const int m_wg = blockIdx.y * 128;
  const int n0 = blockIdx.x * 32;

  // weights: contiguous pre-swizzled chunk, plain b128 copy
  const uint4* srcA = reinterpret_cast<const uint4*>(Wsw + (size_t)blockIdx.y * 16384);
  uint4* dstA = reinterpret_cast<uint4*>(lA);
  for (int e = tid; e < 2048; e += 256) dstA[e] = srcA[e];
  // activations: convert + swizzle
  for (int e = tid; e < 128 * 32; e += 256) {
    int k = e >> 5, n = e & 31;
    lB[((((k >> 5) << 1) + (n >> 4)) << 9) + slotB(k & 31, n & 15)] =
        f2bf(X[((size_t)b * 128 + k) * T_LEN + n0 + n]);
  }
  __syncthreads();

  const int wv = tid >> 5, lane = tid & 31;
  const v16bf* Af = reinterpret_cast<const v16bf*>(lA);
  const v16bf* Bf = reinterpret_cast<const v16bf*>(lB);
  v8f acc0 = {0.f,0.f,0.f,0.f,0.f,0.f,0.f,0.f};
  v8f acc1 = acc0;
#pragma unroll
  for (int kt = 0; kt < 4; ++kt) {
    v16bf a  = Af[(((wv << 2) + kt) << 5) + lane];
    v16bf b0 = Bf[(((kt << 1) + 0) << 5) + lane];
    v16bf b1 = Bf[(((kt << 1) + 1) << 5) + lane];
    acc0 = wmma_bf(a, b0, acc0);
    acc1 = wmma_bf(a, b1, acc1);
  }

  const float alpha = *p1;
  const int lhi = (lane >> 4) << 3;
  const int n_a = n0 + (lane & 15);
  float s = 0.f, s2 = 0.f;
#pragma unroll
  for (int v = 0; v < 8; ++v) {
    int m = m_wg + (wv << 4) + lhi + v;
    float bia = bias[m];
    float y0 = acc0[v] + bia; y0 = y0 >= 0.f ? y0 : alpha * y0;
    float y1 = acc1[v] + bia; y1 = y1 >= 0.f ? y1 : alpha * y1;
    size_t base = ((size_t)b * 512 + m) * T_LEN;
    Z[base + n_a] = y0;
    Z[base + n_a + 16] = y1;
    s += y0 + y1; s2 += y0 * y0 + y1 * y1;
  }
  red2_atomic(s, s2, red, &stats[b * 2], &stats[b * 2 + 1]);
}

// ---------- kernel: GN1(affine) -> dilated depthwise conv -> prelu -> GN2 stats ----------

__global__ __launch_bounds__(256) void dwconv_gn(
    const float* __restrict__ Z1, const float* __restrict__ dw_w,
    const float* __restrict__ dw_b, const float* __restrict__ p2,
    const float* __restrict__ g1, const float* __restrict__ b1,
    const float* __restrict__ statsIn, float* __restrict__ Z2,
    float* __restrict__ statsOut, int dil) {
  __shared__ float red[512];
  const int tid = threadIdx.x;
  const int b = blockIdx.z, h = blockIdx.y;
  const int t = blockIdx.x * 256 + tid;
  const float inv = 1.0f / (512.0f * (float)T_LEN);
  const float mean = statsIn[b * 2] * inv;
  const float rstd = rsqrtf(statsIn[b * 2 + 1] * inv - mean * mean + EPSF);
  const float g = g1[h] * rstd;
  const float bb = b1[h] - mean * g;           // norm(x) = x*g + bb
  const float w0 = dw_w[h * 3], w1 = dw_w[h * 3 + 1], w2 = dw_w[h * 3 + 2];
  const float db = dw_b[h], alpha = *p2;
  float val = 0.f, val2 = 0.f;
  if (t < T_LEN) {
    const float* row = Z1 + ((size_t)b * 512 + h) * T_LEN;
    float acc = db + (row[t] * g + bb) * w1;
    if (t >= dil)          acc += (row[t - dil] * g + bb) * w0;
    if (t + dil < T_LEN)   acc += (row[t + dil] * g + bb) * w2;
    val = acc >= 0.f ? acc : alpha * acc;
    Z2[((size_t)b * 512 + h) * T_LEN + t] = val;
    val2 = val * val;
  }
  red2_atomic(val, val2, red, &statsOut[b * 2], &statsOut[b * 2 + 1]);
}

// ---------- kernel: dual GEMM res/skip over GN2-normalized Z2, accumulate ----------

__global__ __launch_bounds__(256) void gemm_res_sk(
    const float* __restrict__ Z2, const float* __restrict__ statsIn,
    const float* __restrict__ g2, const float* __restrict__ b2,
    const unsigned short* __restrict__ WrSw, const float* __restrict__ br,
    const unsigned short* __restrict__ WsSw, const float* __restrict__ bsk,
    float* __restrict__ OUT, float* __restrict__ SKIP) {
  __shared__ alignas(32) unsigned short lAr[128 * 64];  // [mtile8][kt2][512]
  __shared__ alignas(32) unsigned short lAs[128 * 64];
  __shared__ alignas(32) unsigned short lB[64 * 32];    // [kt2][ntile2][512]

  const int tid = threadIdx.x;
  const int b = blockIdx.z;
  const int n0 = blockIdx.x * 32;
  const float inv = 1.0f / (512.0f * (float)T_LEN);
  const float mean = statsIn[b * 2] * inv;
  const float rstd = rsqrtf(statsIn[b * 2 + 1] * inv - mean * mean + EPSF);

  const int wv = tid >> 5, lane = tid & 31;
  v8f ar0 = {0.f,0.f,0.f,0.f,0.f,0.f,0.f,0.f};
  v8f ar1 = ar0, as0 = ar0, as1 = ar0;

  for (int kc = 0; kc < 8; ++kc) {
    const int kbase = kc << 6;
    if (kc < 7) {  // prefetch next pre-swizzled weight chunk (global_prefetch_b8)
      __builtin_prefetch(&WrSw[(size_t)(kc + 1) * 8192 + tid * 32], 0, 0);
      __builtin_prefetch(&WsSw[(size_t)(kc + 1) * 8192 + tid * 32], 0, 0);
    }
    {
      const uint4* srcR = reinterpret_cast<const uint4*>(WrSw + (size_t)kc * 8192);
      const uint4* srcS = reinterpret_cast<const uint4*>(WsSw + (size_t)kc * 8192);
      uint4* dR = reinterpret_cast<uint4*>(lAr);
      uint4* dS = reinterpret_cast<uint4*>(lAs);
      for (int e = tid; e < 1024; e += 256) { dR[e] = srcR[e]; dS[e] = srcS[e]; }
    }
    for (int e = tid; e < 64 * 32; e += 256) {
      int k = e >> 5, n = e & 31;
      int kg = kbase + k;
      float x = Z2[((size_t)b * 512 + kg) * T_LEN + n0 + n];
      x = (x - mean) * rstd * g2[kg] + b2[kg];
      lB[((((k >> 5) << 1) + (n >> 4)) << 9) + slotB(k & 31, n & 15)] = f2bf(x);
    }
    __syncthreads();
    const v16bf* Arf = reinterpret_cast<const v16bf*>(lAr);
    const v16bf* Asf = reinterpret_cast<const v16bf*>(lAs);
    const v16bf* Bf  = reinterpret_cast<const v16bf*>(lB);
#pragma unroll
    for (int kt = 0; kt < 2; ++kt) {
      v16bf a_r = Arf[(((wv << 1) + kt) << 5) + lane];
      v16bf a_s = Asf[(((wv << 1) + kt) << 5) + lane];
      v16bf b0  = Bf[(((kt << 1) + 0) << 5) + lane];
      v16bf b1  = Bf[(((kt << 1) + 1) << 5) + lane];
      ar0 = wmma_bf(a_r, b0, ar0);
      ar1 = wmma_bf(a_r, b1, ar1);
      as0 = wmma_bf(a_s, b0, as0);
      as1 = wmma_bf(a_s, b1, as1);
    }
    __syncthreads();
  }

  const int lhi = (lane >> 4) << 3;
  const int n_a = n0 + (lane & 15);
#pragma unroll
  for (int v = 0; v < 8; ++v) {
    int m = (wv << 4) + lhi + v;
    size_t o = ((size_t)b * 128 + m) * T_LEN + n_a;
    float rb = br[m], sb = bsk[m];
    OUT[o]        += ar0[v] + rb;
    OUT[o + 16]   += ar1[v] + rb;
    SKIP[o]       += as0[v] + sb;
    SKIP[o + 16]  += as1[v] + sb;
  }
}

// ---------- kernel: generic GEMM with input transform (GN-affine or PReLU) ----------
// mode 0: B-in = (x-mean)*rstd*gvec[k] + bvec[k]   (bottleneck)
// mode 1: B-in = prelu(x, *pscal)                   (heads)

__global__ __launch_bounds__(256) void gemm_bx(
    const float* __restrict__ Bsrc, int kdim,
    const unsigned short* __restrict__ Wsw, const float* __restrict__ bias,
    int mode, const float* __restrict__ stats,
    const float* __restrict__ gvec, const float* __restrict__ bvec,
    const float* __restrict__ pscal,
    float* __restrict__ OUTp, int mtot) {
  __shared__ alignas(32) unsigned short lA[128 * 128];
  __shared__ alignas(32) unsigned short lB[128 * 32];
  const int tid = threadIdx.x;
  const int b = blockIdx.z;
  const int m_wg = blockIdx.y * 128;
  const int n0 = blockIdx.x * 32;

  float mean = 0.f, rstd = 0.f, alpha = 0.f;
  if (mode == 0) {
    float inv = 1.0f / ((float)kdim * (float)T_LEN);
    mean = stats[b * 2] * inv;
    rstd = rsqrtf(stats[b * 2 + 1] * inv - mean * mean + EPSF);
  } else {
    alpha = *pscal;
  }

  const int wv = tid >> 5, lane = tid & 31;
  v8f acc0 = {0.f,0.f,0.f,0.f,0.f,0.f,0.f,0.f};
  v8f acc1 = acc0;
  const int nchunk = kdim >> 7;
  for (int kc = 0; kc < nchunk; ++kc) {
    const int kbase = kc << 7;
    if (kc + 1 < nchunk)
      __builtin_prefetch(&Wsw[((size_t)blockIdx.y * nchunk + kc + 1) * 16384 + tid * 64], 0, 0);
    {
      const uint4* srcA = reinterpret_cast<const uint4*>(
          Wsw + ((size_t)blockIdx.y * nchunk + kc) * 16384);
      uint4* dstA = reinterpret_cast<uint4*>(lA);
      for (int e = tid; e < 2048; e += 256) dstA[e] = srcA[e];
    }
    for (int e = tid; e < 128 * 32; e += 256) {
      int k = e >> 5, n = e & 31;
      int kg = kbase + k;
      float x = Bsrc[((size_t)b * kdim + kg) * T_LEN + n0 + n];
      x = (mode == 0) ? ((x - mean) * rstd * gvec[kg] + bvec[kg])
                      : (x >= 0.f ? x : alpha * x);
      lB[((((k >> 5) << 1) + (n >> 4)) << 9) + slotB(k & 31, n & 15)] = f2bf(x);
    }
    __syncthreads();
    const v16bf* Af = reinterpret_cast<const v16bf*>(lA);
    const v16bf* Bf = reinterpret_cast<const v16bf*>(lB);
#pragma unroll
    for (int kt = 0; kt < 4; ++kt) {
      v16bf a  = Af[(((wv << 2) + kt) << 5) + lane];
      v16bf b0 = Bf[(((kt << 1) + 0) << 5) + lane];
      v16bf b1 = Bf[(((kt << 1) + 1) << 5) + lane];
      acc0 = wmma_bf(a, b0, acc0);
      acc1 = wmma_bf(a, b1, acc1);
    }
    __syncthreads();
  }

  const int lhi = (lane >> 4) << 3;
  const int n_a = n0 + (lane & 15);
#pragma unroll
  for (int v = 0; v < 8; ++v) {
    int m = m_wg + (wv << 4) + lhi + v;
    float bia = bias[m];
    size_t o = ((size_t)b * mtot + m) * T_LEN + n_a;
    OUTp[o]      = acc0[v] + bia;
    OUTp[o + 16] = acc1[v] + bia;
  }
}

// ---------- host orchestration ----------

extern "C" void kernel_launch(void* const* d_in, const int* in_sizes, int n_in,
                              void* d_out, int out_size, void* d_ws, size_t ws_size,
                              hipStream_t stream) {
  (void)in_sizes; (void)n_in; (void)out_size; (void)ws_size;
  const float* input = (const float*)d_in[0];
  const float* ln1_g = (const float*)d_in[1];
  const float* ln1_b = (const float*)d_in[2];
  const float* bn1_w = (const float*)d_in[3];
  const float* bn1_b = (const float*)d_in[4];
  const float* pw_w  = (const float*)d_in[5];
  const float* pw_b  = (const float*)d_in[6];
  const float* dw_w  = (const float*)d_in[7];
  const float* dw_b  = (const float*)d_in[8];
  const float* res_w = (const float*)d_in[9];
  const float* res_b = (const float*)d_in[10];
  const float* sk_w  = (const float*)d_in[11];
  const float* sk_b  = (const float*)d_in[12];
  const float* p1    = (const float*)d_in[13];
  const float* p2    = (const float*)d_in[14];
  const float* n1g   = (const float*)d_in[15];
  const float* n1b   = (const float*)d_in[16];
  const float* n2g   = (const float*)d_in[17];
  const float* n2b   = (const float*)d_in[18];
  const float* h1_p  = (const float*)d_in[19];
  const float* h1_w  = (const float*)d_in[20];
  const float* h1_b  = (const float*)d_in[21];
  const float* h2_p  = (const float*)d_in[22];
  const float* h2_w  = (const float*)d_in[23];
  const float* h2_b  = (const float*)d_in[24];
  float* out = (float*)d_out;

  float* ws = (float*)d_ws;
  const size_t ZSZ = (size_t)4 * 512 * T_LEN;   // 8,192,000 floats
  const size_t OSZ = (size_t)4 * 128 * T_LEN;   // 2,048,000 floats
  float* z1   = ws;
  float* z2   = z1 + ZSZ;
  float* cur  = z2 + ZSZ;
  float* skp  = cur + OSZ;
  float* cur2 = skp + OSZ;
  float* skp2 = cur2 + OSZ;
  float* st1  = skp2 + OSZ;        // [32][4][2]
  float* st2  = st1 + 32 * 8;      // [32][4][2]
  float* st0  = st2 + 32 * 8;      // [4][2]
  // pre-swizzled bf16 weights (16-byte aligned: 24,576,520 floats * 4 % 16 == 0)
  unsigned short* pw_sw  = (unsigned short*)(st0 + 8);
  unsigned short* res_sw = pw_sw  + (size_t)32 * 512 * 128;
  unsigned short* sk_sw  = res_sw + (size_t)32 * 128 * 512;
  unsigned short* bn1_sw = sk_sw  + (size_t)32 * 128 * 512;
  unsigned short* h1_sw  = bn1_sw + 128 * 256;
  unsigned short* h2_sw  = h1_sw  + 256 * 128;

  hipMemsetAsync(skp, 0, OSZ * sizeof(float), stream);
  hipMemsetAsync(st1, 0, (32 * 8 + 32 * 8 + 8) * sizeof(float), stream);

  dim3 blk(256);

  // one-shot weight pre-swizzle into bf16 fragment layout
  prep_w<<<dim3((32 * 512 * 128 + 255) / 256), blk, 0, stream>>>(pw_w,  pw_sw,  32 * 512, 128, 128);
  prep_w<<<dim3((32 * 128 * 512 + 255) / 256), blk, 0, stream>>>(res_w, res_sw, 32 * 128, 512, 64);
  prep_w<<<dim3((32 * 128 * 512 + 255) / 256), blk, 0, stream>>>(sk_w,  sk_sw,  32 * 128, 512, 64);
  prep_w<<<dim3((128 * 256 + 255) / 256),      blk, 0, stream>>>(bn1_w, bn1_sw, 128, 256, 128);
  prep_w<<<dim3((256 * 128 + 255) / 256),      blk, 0, stream>>>(h1_w,  h1_sw,  256, 128, 128);
  prep_w<<<dim3((256 * 128 + 255) / 256),      blk, 0, stream>>>(h2_w,  h2_sw,  256, 128, 128);

  // prologue: GN(input) -> bottleneck GEMM
  stats_sum<<<dim3(64, 1, 4), blk, 0, stream>>>(input, 256 * T_LEN, st0);
  gemm_bx<<<dim3(125, 1, 4), blk, 0, stream>>>(input, 256, bn1_sw, bn1_b, 0, st0,
                                               ln1_g, ln1_b, nullptr, cur, 128);

  auto run_block = [&](int i, float* o, float* s) {
    gemm_pw<<<dim3(125, 4, 4), blk, 0, stream>>>(
        o, pw_sw + (size_t)i * 4 * 16384, pw_b + i * 512, p1 + i, z1, st1 + i * 8);
    dwconv_gn<<<dim3(16, 512, 4), blk, 0, stream>>>(
        z1, dw_w + i * 512 * 3, dw_b + i * 512, p2 + i,
        n1g + i * 512, n1b + i * 512, st1 + i * 8, z2, st2 + i * 8, 1 << (i & 7));
    gemm_res_sk<<<dim3(125, 1, 4), blk, 0, stream>>>(
        z2, st2 + i * 8, n2g + i * 512, n2b + i * 512,
        res_sw + (size_t)i * 65536, res_b + i * 128,
        sk_sw + (size_t)i * 65536, sk_b + i * 128, o, s);
  };

  for (int i = 0; i < 8; ++i) run_block(i, cur, skp);
  // feature_r = out after block 7 (third output, after out1+out2)
  hipMemcpyAsync(out + (size_t)2 * 4 * 256 * T_LEN, cur, OSZ * sizeof(float),
                 hipMemcpyDeviceToDevice, stream);
  for (int i = 8; i < 16; ++i) run_block(i, cur, skp);
  // fork the two output branches
  hipMemcpyAsync(cur2, cur, OSZ * sizeof(float), hipMemcpyDeviceToDevice, stream);
  hipMemcpyAsync(skp2, skp, OSZ * sizeof(float), hipMemcpyDeviceToDevice, stream);
  for (int i = 16; i < 24; ++i) run_block(i, cur, skp);
  for (int i = 24; i < 32; ++i) run_block(i, cur2, skp2);

  // heads: out = h_w @ prelu(skip) + h_b
  gemm_bx<<<dim3(125, 2, 4), blk, 0, stream>>>(skp, 128, h1_sw, h1_b, 1, nullptr,
                                               nullptr, nullptr, h1_p, out, 256);
  gemm_bx<<<dim3(125, 2, 4), blk, 0, stream>>>(skp2, 128, h2_sw, h2_b, 1, nullptr,
                                               nullptr, nullptr, h2_p,
                                               out + (size_t)4 * 256 * T_LEN, 256);
}
